// CharCNN_30159260352734
// MI455X (gfx1250) — matrix operand order repaired and checked by reading, not compiled, not run on hardware
//
#include <hip/hip_runtime.h>
#include <hip/hip_bf16.h>
#include <math.h>

typedef __bf16 bf16;
typedef __attribute__((ext_vector_type(16))) __bf16 v16bf;
typedef __attribute__((ext_vector_type(8)))  __bf16 v8bf;
typedef __attribute__((ext_vector_type(8)))  float  v8f;

union AFrag { v16bf v; v8bf h[2]; };

__device__ __forceinline__ float leakyf(float x) { return x >= 0.f ? x : 0.01f * x; }
__device__ __forceinline__ float sigmoidf_(float x) { return 1.f / (1.f + __expf(-x)); }

// ---------------------------------------------------------------------------
// Prep: build padded conv weight matrix WcT[480][128] (col-major of B, i.e.
// per-output-channel K-contiguous) + concatenated conv bias[480].
// Conv weight layout per reference: [O, 16, K] row-major.
// Row q of the conceptual B: q = j*16 + c  (tap j, channel c); j >= K -> 0.
// ---------------------------------------------------------------------------
__global__ __launch_bounds__(256)
void k_build_wct(const float* __restrict__ w1, const float* __restrict__ w2,
                 const float* __restrict__ w3, const float* __restrict__ w4,
                 const float* __restrict__ w5, const float* __restrict__ w6,
                 const float* __restrict__ w7,
                 const float* __restrict__ b1, const float* __restrict__ b2,
                 const float* __restrict__ b3, const float* __restrict__ b4,
                 const float* __restrict__ b5, const float* __restrict__ b6,
                 const float* __restrict__ b7,
                 bf16* __restrict__ WcT, float* __restrict__ biasc)
{
    int i = blockIdx.x * 256 + threadIdx.x;
    if (i >= 480 * 128) return;
    int o = i >> 7;          // output channel 0..479
    int q = i & 127;         // padded K index 0..127
    int j = q >> 4, c = q & 15;

    int base, K; const float* w; const float* bb;
    if      (o <  32) { base = 0;   K = 1; w = w1; bb = b1; }
    else if (o <  64) { base = 32;  K = 2; w = w2; bb = b2; }
    else if (o <  96) { base = 64;  K = 3; w = w3; bb = b3; }
    else if (o < 160) { base = 96;  K = 4; w = w4; bb = b4; }
    else if (o < 224) { base = 160; K = 5; w = w5; bb = b5; }
    else if (o < 352) { base = 224; K = 6; w = w6; bb = b6; }
    else              { base = 352; K = 7; w = w7; bb = b7; }
    int ol = o - base;
    float val = (j < K) ? w[(ol * 16 + c) * K + j] : 0.f;
    WcT[o * 128 + q] = (bf16)val;
    if (q == 0) biasc[o] = bb[ol];
}

__global__ __launch_bounds__(256)
void k_f32_to_bf16(const float* __restrict__ src, bf16* __restrict__ dst, int n)
{
    int i = blockIdx.x * 256 + threadIdx.x;
    if (i < n) dst[i] = (bf16)src[i];
}

// ---------------------------------------------------------------------------
// Embed + all 7 convs as ONE Toeplitz GEMM per token + masked max-pool + leaky.
// One wave per token; E[72][16] bf16 staged in LDS (rows >=50 zero so padded
// weight taps hit zeros). A[p][q] = E_flat[16p + q]  (free im2col).
// GEMM: M=64 (4 tiles, positions), K=128 (4 steps), N=480 (30 tiles).
// ---------------------------------------------------------------------------
#define TOKENS_PER_BLOCK 8
#define E_ROWS 72

__global__ __launch_bounds__(256)
void k_embed_conv(const int* __restrict__ X, const float* __restrict__ emb,
                  const bf16* __restrict__ WcT, const float* __restrict__ biasc,
                  bf16* __restrict__ H)
{
    __shared__ __align__(32) bf16 Es[TOKENS_PER_BLOCK * E_ROWS * 16];
    const int lane  = threadIdx.x & 31;
    const int wv    = threadIdx.x >> 5;
    const int token = blockIdx.x * TOKENS_PER_BLOCK + wv;
    bf16* E = &Es[wv * E_ROWS * 16];

    // Gather char embeddings -> LDS (bf16); zero-pad rows 50..71.
    for (int i = lane; i < E_ROWS; i += 32) {
        if (i < 50) {
            int idx = X[token * 50 + i];
            const float* er = emb + idx * 16;
#pragma unroll
            for (int c = 0; c < 16; ++c) E[i * 16 + c] = (bf16)er[c];
        } else {
#pragma unroll
            for (int c = 0; c < 16; ++c) E[i * 16 + c] = (bf16)0.f;
        }
    }
    __syncthreads();

    const int hi = lane >> 4;   // half-wave select
    const int ln = lane & 15;

    for (int nt = 0; nt < 30; nt += 2) {
        v8f acc[4][2];
#pragma unroll
        for (int mt = 0; mt < 4; ++mt) { acc[mt][0] = (v8f)(0.f); acc[mt][1] = (v8f)(0.f); }

#pragma unroll
        for (int kq = 0; kq < 4; ++kq) {
            const int k0 = kq * 32;
            AFrag a[4];
#pragma unroll
            for (int mt = 0; mt < 4; ++mt) {
                int p = mt * 16 + ln;                       // position row
                const bf16* base = E + p * 16 + k0 + hi * 16; // Toeplitz view
                a[mt].h[0] = *(const v8bf*)(base);
                a[mt].h[1] = *(const v8bf*)(base + 8);
            }
#pragma unroll
            for (int j = 0; j < 2; ++j) {
                int o = (nt + j) * 16 + ln;
                v16bf b = *(const v16bf*)(WcT + o * 128 + k0 + hi * 16);
#pragma unroll
                for (int mt = 0; mt < 4; ++mt)
                    acc[mt][j] = __builtin_amdgcn_wmma_f32_16x16x32_bf16(
                        false, a[mt].v, false, b, (short)0, acc[mt][j], false, false);
            }
        }

        // Masked max over positions, + bias, leaky, store h0 (bf16).
#pragma unroll
        for (int j = 0; j < 2; ++j) {
            int tile = nt + j;
            // filter width of this 16-channel tile (group boundaries align to tiles)
            int kw = 1 + (tile >= 2) + (tile >= 4) + (tile >= 6) +
                         (tile >= 10) + (tile >= 14) + (tile >= 22);
            int pmax = 50 - kw;
            float m = -3.0e38f;
#pragma unroll
            for (int mt = 0; mt < 4; ++mt)
#pragma unroll
                for (int r = 0; r < 8; ++r) {
                    int p = mt * 16 + r + hi * 8;   // D layout: M = r + 8*(lane>=16)
                    if (p <= pmax) m = fmaxf(m, acc[mt][j][r]);
                }
            m = fmaxf(m, __shfl_xor(m, 16, 32));    // combine the two half-waves
            int o = tile * 16 + ln;
            float res = leakyf(m + biasc[o]);
            if (hi == 0) H[(size_t)token * 480 + o] = (bf16)res;
        }
    }
}

// ---------------------------------------------------------------------------
// Highway layer: proj = x @ W^T + b ; (nonlin, gate) = split(proj);
// y = sigmoid(gate)*x + (1-sigmoid(gate))*leaky(nonlin).
// Each wave: 64x16 output tile of the 480-wide result, with TWO accumulator
// sets (cols c and c+480) so the gating fuses in-register. K = 480 = 15 steps.
// ---------------------------------------------------------------------------
__global__ __launch_bounds__(256)
void k_highway(const bf16* __restrict__ Xbf, const bf16* __restrict__ Wbf,
               const float* __restrict__ bias, bf16* __restrict__ Ybf)
{
    const int lane = threadIdx.x & 31;
    const int wv   = threadIdx.x >> 5;
    const int task = blockIdx.x * 8 + wv;
    const int mg = task / 30;            // 512 M-groups of 64 rows
    const int ct = task % 30;            // 30 col tiles of 16
    const int hi = lane >> 4, ln = lane & 15;
    const int row0 = mg * 64;
    const int colg = ct * 16 + ln;

    v8f accN[4], accG[4];
#pragma unroll
    for (int mt = 0; mt < 4; ++mt) { accN[mt] = (v8f)(0.f); accG[mt] = (v8f)(0.f); }

    for (int kk = 0; kk < 15; ++kk) {
        const int k0 = kk * 32;
        AFrag a[4];
#pragma unroll
        for (int mt = 0; mt < 4; ++mt) {
            int row = row0 + mt * 16 + ln;
            const bf16* base = Xbf + (size_t)row * 480 + k0 + hi * 16;
            a[mt].h[0] = *(const v8bf*)(base);
            a[mt].h[1] = *(const v8bf*)(base + 8);
        }
        v16bf bn = *(const v16bf*)(Wbf + (size_t)colg * 480 + k0 + hi * 16);
        v16bf bg = *(const v16bf*)(Wbf + (size_t)(colg + 480) * 480 + k0 + hi * 16);
#pragma unroll
        for (int mt = 0; mt < 4; ++mt) {
            accN[mt] = __builtin_amdgcn_wmma_f32_16x16x32_bf16(
                false, a[mt].v, false, bn, (short)0, accN[mt], false, false);
            accG[mt] = __builtin_amdgcn_wmma_f32_16x16x32_bf16(
                false, a[mt].v, false, bg, (short)0, accG[mt], false, false);
        }
    }

    const float bn = bias[colg], bg = bias[colg + 480];
#pragma unroll
    for (int mt = 0; mt < 4; ++mt)
#pragma unroll
        for (int r = 0; r < 8; ++r) {
            int row = row0 + mt * 16 + r + hi * 8;
            float xv = (float)Xbf[(size_t)row * 480 + colg];
            float g  = sigmoidf_(accG[mt][r] + bg);
            float nl = leakyf(accN[mt][r] + bn);
            Ybf[(size_t)row * 480 + colg] = (bf16)(g * xv + (1.f - g) * nl);
        }
}

// ---------------------------------------------------------------------------
// Output projection: out = h @ proj_w^T + proj_b  ([32768,480]x[480,768], f32 out)
// ---------------------------------------------------------------------------
__global__ __launch_bounds__(256)
void k_proj(const bf16* __restrict__ Xbf, const bf16* __restrict__ Wbf,
            const float* __restrict__ bias, float* __restrict__ Out)
{
    const int lane = threadIdx.x & 31;
    const int wv   = threadIdx.x >> 5;
    const int task = blockIdx.x * 8 + wv;
    const int mg = task / 48;
    const int ct = task % 48;
    const int hi = lane >> 4, ln = lane & 15;
    const int row0 = mg * 64;
    const int colg = ct * 16 + ln;

    v8f acc[4];
#pragma unroll
    for (int mt = 0; mt < 4; ++mt) acc[mt] = (v8f)(0.f);

    for (int kk = 0; kk < 15; ++kk) {
        const int k0 = kk * 32;
        AFrag a[4];
#pragma unroll
        for (int mt = 0; mt < 4; ++mt) {
            int row = row0 + mt * 16 + ln;
            const bf16* base = Xbf + (size_t)row * 480 + k0 + hi * 16;
            a[mt].h[0] = *(const v8bf*)(base);
            a[mt].h[1] = *(const v8bf*)(base + 8);
        }
        v16bf b = *(const v16bf*)(Wbf + (size_t)colg * 480 + k0 + hi * 16);
#pragma unroll
        for (int mt = 0; mt < 4; ++mt)
            acc[mt] = __builtin_amdgcn_wmma_f32_16x16x32_bf16(
                false, a[mt].v, false, b, (short)0, acc[mt], false, false);
    }

    const float bv = bias[colg];
#pragma unroll
    for (int mt = 0; mt < 4; ++mt)
#pragma unroll
        for (int r = 0; r < 8; ++r) {
            int row = row0 + mt * 16 + r + hi * 8;
            Out[(size_t)row * 768 + colg] = acc[mt][r] + bv;
        }
}

// ---------------------------------------------------------------------------
extern "C" void kernel_launch(void* const* d_in, const int* in_sizes, int n_in,
                              void* d_out, int out_size, void* d_ws, size_t ws_size,
                              hipStream_t stream)
{
    (void)in_sizes; (void)n_in; (void)out_size; (void)ws_size;

    const int*   X     = (const int*)d_in[0];
    const float* emb   = (const float*)d_in[1];
    const float* cw[7], *cb[7];
    for (int i = 0; i < 7; ++i) { cw[i] = (const float*)d_in[2 + 2*i]; cb[i] = (const float*)d_in[3 + 2*i]; }
    const float* hw1_w = (const float*)d_in[16];
    const float* hw1_b = (const float*)d_in[17];
    const float* hw2_w = (const float*)d_in[18];
    const float* hw2_b = (const float*)d_in[19];
    const float* pj_w  = (const float*)d_in[20];
    const float* pj_b  = (const float*)d_in[21];

    const int N = 64 * 512;   // 32768 tokens

    // Workspace layout (256B aligned slabs)
    char* ws = (char*)d_ws;
    size_t off = 0;
    auto slab = [&](size_t bytes) { char* p = ws + off; off += (bytes + 255) & ~size_t(255); return p; };
    bf16*  WcT    = (bf16*) slab(480 * 128 * sizeof(bf16));       // padded conv weights
    float* biasc  = (float*)slab(480 * sizeof(float));            // concat conv bias
    bf16*  hw1bf  = (bf16*) slab(960 * 480 * sizeof(bf16));
    bf16*  hw2bf  = (bf16*) slab(960 * 480 * sizeof(bf16));
    bf16*  pjbf   = (bf16*) slab(768 * 480 * sizeof(bf16));
    bf16*  hA     = (bf16*) slab((size_t)N * 480 * sizeof(bf16)); // ping
    bf16*  hB     = (bf16*) slab((size_t)N * 480 * sizeof(bf16)); // pong

    // 1) weight prep (tiny)
    k_build_wct<<<(480 * 128 + 255) / 256, 256, 0, stream>>>(
        cw[0], cw[1], cw[2], cw[3], cw[4], cw[5], cw[6],
        cb[0], cb[1], cb[2], cb[3], cb[4], cb[5], cb[6], WcT, biasc);
    k_f32_to_bf16<<<(960 * 480 + 255) / 256, 256, 0, stream>>>(hw1_w, hw1bf, 960 * 480);
    k_f32_to_bf16<<<(960 * 480 + 255) / 256, 256, 0, stream>>>(hw2_w, hw2bf, 960 * 480);
    k_f32_to_bf16<<<(768 * 480 + 255) / 256, 256, 0, stream>>>(pj_w,  pjbf,  768 * 480);

    // 2) embed + conv-as-GEMM + maxpool  (1 wave per token)
    k_embed_conv<<<N / TOKENS_PER_BLOCK, 256, 0, stream>>>(X, emb, WcT, biasc, hA);

    // 3) highway x2 (ping-pong)
    k_highway<<<(N / 64) * 30 / 8, 256, 0, stream>>>(hA, hw1bf, hw1_b, hB);
    k_highway<<<(N / 64) * 30 / 8, 256, 0, stream>>>(hB, hw2bf, hw2_b, hA);

    // 4) projection -> f32 output [64,512,768]
    k_proj<<<(N / 64) * 48 / 8, 256, 0, stream>>>(hA, pjbf, pj_b, (float*)d_out);
}
